// RNN_71090298683912
// MI455X (gfx1250) — matrix operand order repaired and checked by reading, not compiled, not run on hardware
//
#include <hip/hip_runtime.h>
#include <hip/hip_bf16.h>
#include <math.h>

// Stacked tanh RNN, persistent pipelined-wavefront kernel for MI455X (gfx1250):
//  - 128 workgroups x 256 threads (8 wave32s).  wg = (nblock, m_tile):
//      nblock in [0,64): owns 16 output columns; m_tile in {0,1}: owns 16 batch rows.
//  - WEIGHTS LIVE IN VGPRs: each wave's B-fragment slice (16 cols x 256 K x 2 layers
//    bf16 = 128 VGPRs) converted from fp32 ONCE into register arrays; every WMMA
//    consumes B straight from registers.  LDS holds only two 8KB reduction buffers.
//  - X PIPELINED TO BF16: grid has exactly BH=32768 threads, so during tick u each
//    thread converts ONE element of x(u+1) into a 2-slot bf16 ring in workspace
//    (prep kernel seeds x(0)).  The hot loop is then uniformly bf16: A fragments are
//    two 16B loads, no cvt, no WMMA WAR nops, and x fp32 is read once per element
//    total instead of 64x per step.
//  - Software pipeline across layers: tick u computes L0(t=u) and L1(t=u-1), which
//    are mutually independent -> ONE grid barrier per tick (2049 total), 16 WMMAs
//    per wave per barrier interval, issued as load-all-fragments then WMMA chain.
//  - NOTE: the single generic tick body with runtime parity selects is deliberate:
//    peeling + parity-unrolling the loop (tried) blows past the 256-VGPR window and
//    makes LLVM rematerialize the weight loads inside the loop.
//  - Hidden state ping-pong buffered in workspace as bf16; K (=2048 concat) split
//    across 8 waves; K-partials reduced via two LDS buffers (one local
//    __syncthreads per tick), then bias + tanhf; bf16 state write + fp32
//    history/h_T writes straight from registers.
//  - Monotonic atomic-counter grid barrier, reset by prep kernel each launch
//    (deterministic under graph capture/replay).

typedef __attribute__((ext_vector_type(16))) __bf16 v16bf;
typedef __attribute__((ext_vector_type(8)))  __bf16 v8bf;
typedef __attribute__((ext_vector_type(8)))  float  v8f;

#define T_SEQ 2048
#define B_    32
#define H_    1024
#define L_    2
#define BH    (B_*H_)
#define NWG   128
#define REDF  (8*256)           // one reduction buffer: 8 waves x 16x16 fp32 tile

__device__ __forceinline__ void grid_barrier(unsigned* cnt, unsigned target) {
  __syncthreads();
  if (threadIdx.x == 0) {
    __hip_atomic_fetch_add(cnt, 1u, __ATOMIC_ACQ_REL, __HIP_MEMORY_SCOPE_AGENT);
    while (__hip_atomic_load(cnt, __ATOMIC_ACQUIRE, __HIP_MEMORY_SCOPE_AGENT) < target) {
      __builtin_amdgcn_s_sleep(1);
    }
  }
  __syncthreads();
}

// A fragment (16x32 bf16): lane L holds row M = L&15; elems 0..15 -> K = base+hi*16+0..15.
__device__ __forceinline__ v16bf load_a_frag(const __bf16* p) {
  v8bf c0 = *(const v8bf*)p;          // 16B
  v8bf c1 = *(const v8bf*)(p + 8);    // 16B
  return __builtin_shufflevector(c0, c1, 0,1,2,3,4,5,6,7,8,9,10,11,12,13,14,15);
}

// B (weight) fragment: lane holds col N = lane15, 16 contiguous fp32 K values -> bf16.
__device__ __forceinline__ v16bf load_w_frag(const float* p) {
  const float4* q = (const float4*)p;
  float4 f0 = q[0], f1 = q[1], f2 = q[2], f3 = q[3];
  v16bf a;
  a[0]=(__bf16)f0.x;  a[1]=(__bf16)f0.y;  a[2]=(__bf16)f0.z;  a[3]=(__bf16)f0.w;
  a[4]=(__bf16)f1.x;  a[5]=(__bf16)f1.y;  a[6]=(__bf16)f1.z;  a[7]=(__bf16)f1.w;
  a[8]=(__bf16)f2.x;  a[9]=(__bf16)f2.y;  a[10]=(__bf16)f2.z; a[11]=(__bf16)f2.w;
  a[12]=(__bf16)f3.x; a[13]=(__bf16)f3.y; a[14]=(__bf16)f3.z; a[15]=(__bf16)f3.w;
  return a;
}

// One wave's K-chunk: issue A-fragment loads, then WMMAs with register-resident B.
__device__ __forceinline__ v8f mm_chain(const __bf16* __restrict__ arow,
                                        const v16bf* wreg, int hi) {
  v16bf a[8];
#pragma unroll
  for (int s = 0; s < 8; ++s) a[s] = load_a_frag(arow + s * 32 + hi * 16);
  v8f acc = {};
#pragma unroll
  for (int s = 0; s < 8; ++s)
    acc = __builtin_amdgcn_wmma_f32_16x16x32_bf16(
        false, a[s], false, wreg[s], (short)0, acc, false, false);
  return acc;
}

__global__ void rnn_prep(const float* __restrict__ x, const float* __restrict__ h0in,
                         __bf16* __restrict__ xbf0,
                         __bf16* __restrict__ h0b1, __bf16* __restrict__ h1b1,
                         unsigned* __restrict__ bar) {
  int i = blockIdx.x * 256 + threadIdx.x;
  if (i == 0) *bar = 0u;
  if (i < BH) xbf0[i] = (__bf16)x[i];                              // x(0) -> ring slot 0
  if (i < BH)           h0b1[i]      = (__bf16)h0in[i];            // layer-0 init state
  else if (i < L_ * BH) h1b1[i - BH] = (__bf16)h0in[i];            // layer-1 init state
}

__global__ __launch_bounds__(256) void rnn_persist(
    const float* __restrict__ x,
    const float* __restrict__ W_ih, const float* __restrict__ W_hh,
    const float* __restrict__ b_ih, const float* __restrict__ b_hh,
    __bf16* __restrict__ xbf0, __bf16* __restrict__ xbf1,
    __bf16* __restrict__ h0b0, __bf16* __restrict__ h0b1,
    __bf16* __restrict__ h1b0, __bf16* __restrict__ h1b1,
    unsigned* __restrict__ bar, float* __restrict__ out)
{
  __shared__ float red0[REDF];
  __shared__ float red1[REDF];

  const int tid = threadIdx.x;
  const int wave = tid >> 5;
  const int lane = tid & 31;
  const int lane15 = lane & 15;
  const int hi = lane >> 4;
  const int nblock = blockIdx.x >> 1;
  const int wg_m   = blockIdx.x & 1;
  const int ncol_base = nblock * 16;
  const int brow_base = wg_m * 16;

  // --- One-time: load this wave's weight slice into REGISTERS as bf16 B fragments.
  // Wave's concat-K chunk is [wave*256, wave*256+256): waves 0-3 -> W_ih, 4-7 -> W_hh.
  v16bf wreg0[8], wreg1[8];
  {
    const int c = ncol_base + lane15;                 // weight row = output column
    const int kb = wave * 256 + hi * 16;              // concat-K fragment base
    const float* w0 = (wave < 4)
        ? (W_ih + (size_t)c * H_ + kb)                          // layer 0, ih part
        : (W_hh + (size_t)c * H_ + (kb - H_));                  // layer 0, hh part
    const float* w1 = (wave < 4)
        ? (W_ih + ((size_t)H_ + c) * H_ + kb)                   // layer 1, ih part
        : (W_hh + ((size_t)H_ + c) * H_ + (kb - H_));           // layer 1, hh part
#pragma unroll
    for (int s = 0; s < 8; ++s) {
      wreg0[s] = load_w_frag(w0 + s * 32);
      wreg1[s] = load_w_frag(w1 + s * 32);
    }
  }

  // Per-thread constants.
  const int col = ncol_base + (tid & 15);
  const float bias0 = b_ih[col] + b_hh[col];
  const float bias1 = b_ih[H_ + col] + b_hh[H_ + col];
  const int m = tid >> 4;
  const size_t oidx = (size_t)(brow_base + m) * H_ + col;     // output element index
  const int b = brow_base + lane15;                           // A row = batch index
  const size_t aofs = (size_t)b * H_ + (wave & 3) * 256;      // A row + K-chunk offset
  const int gid = blockIdx.x * 256 + tid;                     // x-conversion element

  float* hT0 = out + (size_t)T_SEQ * BH;        // h_T layer 0 slice
  float* hT1 = hT0 + (size_t)BH;                // h_T layer 1 slice

  unsigned target = 0;
  // Tick u: compute L0(t=u) [u<T] and L1(t=u-1) [u>=1]; they are independent.
  for (int u = 0; u <= T_SEQ; ++u) {
    const bool doL0 = (u < T_SEQ);
    const bool doL1 = (u >= 1);
    const __bf16* xcur = (u & 1) ? xbf1 : xbf0; // x(u) bf16, converted last tick
    const __bf16* h0p = (u & 1) ? h0b0 : h0b1;  // h0(u-1)
    __bf16*       h0n = (u & 1) ? h0b1 : h0b0;  // h0(u)
    const __bf16* h1p = (u & 1) ? h1b1 : h1b0;  // h1(u-2)
    __bf16*       h1n = (u & 1) ? h1b0 : h1b1;  // h1(u-1)

    v8f acc0 = {}, acc1 = {};
    if (doL0) {     // L0(u): A = [x_u | h0(u-1)]   (all-bf16 sources)
      acc0 = mm_chain(((wave >= 4) ? h0p : xcur) + aofs, wreg0, hi);
    }
    if (doL1) {     // L1(u-1): A = [h0(u-1) | h1(u-2)]
      acc1 = mm_chain(((wave >= 4) ? h1p : h0p) + aofs, wreg1, hi);
    }

    // Pipeline: convert one element of x(u+1) into the other ring slot.
    if (u + 1 < T_SEQ) {
      __bf16* xnxt = ((u + 1) & 1) ? xbf1 : xbf0;
      xnxt[gid] = (__bf16)x[(size_t)(u + 1) * BH + gid];
    }

    // Cross-wave K reduction: C layout lane L: VGPR r -> M = r + hi*8, N = L&15.
    if (doL0) {
#pragma unroll
      for (int r = 0; r < 8; ++r)
        red0[wave * 256 + (r + hi * 8) * 16 + lane15] = acc0[r];
    }
    if (doL1) {
#pragma unroll
      for (int r = 0; r < 8; ++r)
        red1[wave * 256 + (r + hi * 8) * 16 + lane15] = acc1[r];
    }
    __syncthreads();

    if (doL0) {
      float s = 0.f;
#pragma unroll
      for (int w = 0; w < 8; ++w) s += red0[w * 256 + tid];
      const float hv = tanhf(s + bias0);
      h0n[oidx] = (__bf16)hv;
      if (u == T_SEQ - 1) hT0[oidx] = hv;
    }
    if (doL1) {
      float s = 0.f;
#pragma unroll
      for (int w = 0; w < 8; ++w) s += red1[w * 256 + tid];
      const float hv = tanhf(s + bias1);
      h1n[oidx] = (__bf16)hv;
      out[(size_t)(u - 1) * BH + oidx] = hv;    // history[t = u-1]
      if (u == T_SEQ) hT1[oidx] = hv;
    }

    target += NWG; grid_barrier(bar, target);
  }
}

extern "C" void kernel_launch(void* const* d_in, const int* in_sizes, int n_in,
                              void* d_out, int out_size, void* d_ws, size_t ws_size,
                              hipStream_t stream) {
  (void)in_sizes; (void)n_in; (void)out_size; (void)ws_size;
  const float* x    = (const float*)d_in[0];
  const float* h0   = (const float*)d_in[1];
  const float* W_ih = (const float*)d_in[2];
  const float* W_hh = (const float*)d_in[3];
  const float* b_ih = (const float*)d_in[4];
  const float* b_hh = (const float*)d_in[5];
  float* out = (float*)d_out;

  __bf16* xbf0 = (__bf16*)d_ws;          // [B,H] bf16 x ring (2 slots)
  __bf16* xbf1 = xbf0 + (size_t)BH;
  __bf16* h0b0 = xbf1 + (size_t)BH;      // [B,H] bf16 state buffers (ping/pong x 2 layers)
  __bf16* h0b1 = h0b0 + (size_t)BH;
  __bf16* h1b0 = h0b1 + (size_t)BH;
  __bf16* h1b1 = h1b0 + (size_t)BH;
  unsigned* bar = (unsigned*)(h1b1 + (size_t)BH);

  rnn_prep<<<(L_ * BH + 255) / 256, 256, 0, stream>>>(x, h0, xbf0, h0b1, h1b1, bar);
  rnn_persist<<<NWG, 256, 0, stream>>>(
      x, W_ih, W_hh, b_ih, b_hh, xbf0, xbf1, h0b0, h0b1, h1b0, h1b1, bar, out);
}
